// PointNetFeaturePropagation_2972117368914
// MI455X (gfx1250) — matrix operand order repaired and checked by reading, not compiled, not run on hardware
//
#include <hip/hip_runtime.h>
#include <hip/hip_bf16.h>

typedef __attribute__((ext_vector_type(16))) __bf16 v16bf;
typedef __attribute__((ext_vector_type(8)))  float  v8f;

#define B_   16
#define N_   4096
#define S_   1024
#define D1_  128
#define D2_  256
#define CIN_ 384
#define C0_  256
#define C1_  128
#define EPS_BN 1e-5f

// ---------------------------------------------------------------------------
// float -> bf16, round-to-nearest-even (explicit bit math; target-independent)
// ---------------------------------------------------------------------------
__device__ inline __bf16 f2bf(float f) {
  unsigned u = __builtin_bit_cast(unsigned, f);
  unsigned r = (u + 0x7FFFu + ((u >> 16) & 1u)) >> 16;
  unsigned short h = (unsigned short)r;
  return __builtin_bit_cast(__bf16, h);
}

// ---------------------------------------------------------------------------
// K1: 3-NN + interpolation weights.
//   idx  = indices of 3 smallest distances (stable / smallest-index on ties,
//          matching argsort)
//   wgt  = normalized 1/d over the UNSORTED first 3 columns (s = 0,1,2),
//          faithful to the reference's quirk.
// One block = 256 points of one batch; xyz2[b] (12 KB) staged in LDS.
// ---------------------------------------------------------------------------
__global__ __launch_bounds__(256) void knn_kernel(
    const float* __restrict__ xyz1, const float* __restrict__ xyz2,
    int* __restrict__ idxOut, float* __restrict__ wgtOut)
{
  __shared__ float sx[S_], sy[S_], sz[S_];
  int b = blockIdx.x >> 4;          // N_/256 = 16 chunks per batch
  int chunk = blockIdx.x & 15;
  int tid = threadIdx.x;
  const float* x2b = xyz2 + (size_t)b * 3 * S_;
  for (int i = tid; i < S_; i += 256) {
    sx[i] = x2b[i];
    sy[i] = x2b[S_ + i];
    sz[i] = x2b[2 * S_ + i];
  }
  __syncthreads();

  int n = chunk * 256 + tid;
  const float* x1b = xyz1 + (size_t)b * 3 * N_;
  float qx = x1b[n], qy = x1b[N_ + n], qz = x1b[2 * N_ + n];

  float d0 = 3.4e38f, d1 = 3.4e38f, d2 = 3.4e38f;
  int i0 = 0, i1 = 0, i2 = 0;
  for (int s = 0; s < S_; ++s) {
    float dx = qx - sx[s], dy = qy - sy[s], dz = qz - sz[s];
    float d = dx * dx + dy * dy + dz * dz;
    if (d < d0)      { d2 = d1; i2 = i1; d1 = d0; i1 = i0; d0 = d; i0 = s; }
    else if (d < d1) { d2 = d1; i2 = i1; d1 = d;  i1 = s; }
    else if (d < d2) { d2 = d;  i2 = s; }
  }

  float w[3];
#pragma unroll
  for (int k = 0; k < 3; ++k) {   // distances to s = 0,1,2 (unsorted columns)
    float dx = qx - sx[k], dy = qy - sy[k], dz = qz - sz[k];
    float d = fmaxf(dx * dx + dy * dy + dz * dz, 1e-10f);
    w[k] = 1.0f / d;
  }
  float wsum = w[0] + w[1] + w[2];

  size_t o = ((size_t)b * N_ + n) * 3;
  idxOut[o] = i0; idxOut[o + 1] = i1; idxOut[o + 2] = i2;
  wgtOut[o]     = w[0] / wsum;
  wgtOut[o + 1] = w[1] / wsum;
  wgtOut[o + 2] = w[2] / wsum;
}

// ---------------------------------------------------------------------------
// Pack weights [M x Ktot] f32 row-major -> WMMA A-fragment order (bf16).
// A 16x32 bf16 order used here (contiguous k, B uses the same): lane L holds row M = L%16;
//   element e: K = base0+e, base0 = (L<16?0:16).
// Packed memory: [mtile][ktile][lane][e] -> each lane's frag = 32 contig bytes.
// ---------------------------------------------------------------------------
__global__ void pack_w_kernel(const float* __restrict__ W, __bf16* __restrict__ Wp,
                              int Mtiles, int Ktiles, int Ktot)
{
  int tid = blockIdx.x * blockDim.x + threadIdx.x;   // one lane-slot, writes 16
  int total = Mtiles * Ktiles * 32;
  if (tid >= total) return;
  int lane = tid & 31;
  int kt = (tid >> 5) % Ktiles;
  int mt = (tid >> 5) / Ktiles;
  int m = mt * 16 + (lane & 15);
  int base0 = (lane < 16) ? 0 : 16;
  v16bf frag;
#pragma unroll
  for (int e = 0; e < 16; ++e) {
    int kk = base0 + e;
    frag[e] = f2bf(W[(size_t)m * Ktot + (kt * 32 + kk)]);
  }
  *(v16bf*)(Wp + (size_t)tid * 16) = frag;
}

// ---------------------------------------------------------------------------
// Build GEMM0's B operand (bf16, fragment order) fused with interpolation:
//   channel c < 128  -> points1[b][c][n]
//   channel c >= 128 -> sum_k wgt[k] * points2[b][c-128][idx[k]]
// B 32x16 order used here (contiguous k, same as A): lane L holds column N = L%16; element e: K = (L<16?0:16)+e.
// Packed: [b][ntile][ktile][lane][e].
// ---------------------------------------------------------------------------
__global__ void pack_x0_kernel(const float* __restrict__ p1, const float* __restrict__ p2,
                               const int* __restrict__ idx, const float* __restrict__ wgt,
                               __bf16* __restrict__ Xp)
{
  const int Ntiles = N_ / 16, Ktiles = CIN_ / 32;
  int tid = blockIdx.x * blockDim.x + threadIdx.x;
  int total = B_ * Ntiles * Ktiles * 32;
  if (tid >= total) return;
  int lane = tid & 31;
  int kt = (tid >> 5) % Ktiles;
  int nt = ((tid >> 5) / Ktiles) % Ntiles;
  int b  = (tid >> 5) / (Ktiles * Ntiles);
  int n = nt * 16 + (lane & 15);
  int kbase = kt * 32 + ((lane < 16) ? 0 : 16);

  size_t io = ((size_t)b * N_ + n) * 3;
  int j0 = idx[io], j1 = idx[io + 1], j2 = idx[io + 2];
  float w0 = wgt[io], w1 = wgt[io + 1], w2 = wgt[io + 2];

  v16bf frag;
#pragma unroll
  for (int e = 0; e < 16; ++e) {
    int c = kbase + e;
    float v;
    if (c < D1_) {
      v = p1[((size_t)b * D1_ + c) * N_ + n];
    } else {
      const float* row = p2 + ((size_t)b * D2_ + (c - D1_)) * S_;
      v = w0 * row[j0] + w1 * row[j1] + w2 * row[j2];
    }
    frag[e] = f2bf(v);
  }
  *(v16bf*)(Xp + (size_t)tid * 16) = frag;
}

// ---------------------------------------------------------------------------
// bf16 WMMA GEMM: H[b][M][N_] = A[M x K] * X[b][K x N_], f32 accumulate.
// Wave computes a 16(M) x 64(N) strip; A fragment reused across 4 WMMAs.
// Register double-buffering: k-tile kt+1's fragments are loaded before the
// 4 WMMAs of kt, so the loads overlap compute instead of a full
// s_wait_loadcnt 0 in front of every WMMA.
// Wave mapping is mt-fastest: one 8-wave workgroup covers 8 m-tiles of the
// SAME n-range, so the B-fragment re-reads across m hit L2 back-to-back.
// ---------------------------------------------------------------------------
__global__ __launch_bounds__(256) void gemm_wmma_kernel(
    const __bf16* __restrict__ Ap, const __bf16* __restrict__ Bp,
    float* __restrict__ H, int Mtiles, int Ktiles)
{
  const int Ntiles = N_ / 16;            // 256
  const int NG = Ntiles / 4;             // 64 groups of 4 n-tiles
  int wave = (blockIdx.x * blockDim.x + threadIdx.x) >> 5;
  int lane = threadIdx.x & 31;
  int wavesPerBatch = NG * Mtiles;
  int b  = wave / wavesPerBatch;
  int r  = wave % wavesPerBatch;
  int ng = r / Mtiles;                   // mt fastest within a block
  int mt = r % Mtiles;

  const __bf16* aPtr = Ap + ((size_t)mt * Ktiles * 32 + lane) * 16;
  const __bf16* bPtr = Bp + (((size_t)b * Ntiles + ng * 4) * Ktiles * 32 + lane) * 16;
  const size_t bStride = (size_t)Ktiles * 512;   // elements per n-tile

  v8f a0 = {}, a1 = {}, a2 = {}, a3 = {};

  // prologue: k-tile 0
  v16bf afC  = *(const v16bf*)(aPtr);
  v16bf bf0C = *(const v16bf*)(bPtr);
  v16bf bf1C = *(const v16bf*)(bPtr + bStride);
  v16bf bf2C = *(const v16bf*)(bPtr + 2 * bStride);
  v16bf bf3C = *(const v16bf*)(bPtr + 3 * bStride);

  for (int kt = 0; kt < Ktiles - 1; ++kt) {
    size_t off = (size_t)(kt + 1) * 512;
    v16bf afN  = *(const v16bf*)(aPtr + off);
    v16bf bf0N = *(const v16bf*)(bPtr + off);
    v16bf bf1N = *(const v16bf*)(bPtr + bStride + off);
    v16bf bf2N = *(const v16bf*)(bPtr + 2 * bStride + off);
    v16bf bf3N = *(const v16bf*)(bPtr + 3 * bStride + off);

    a0 = __builtin_amdgcn_wmma_f32_16x16x32_bf16(false, afC, false, bf0C, (short)0, a0, false, false);
    a1 = __builtin_amdgcn_wmma_f32_16x16x32_bf16(false, afC, false, bf1C, (short)0, a1, false, false);
    a2 = __builtin_amdgcn_wmma_f32_16x16x32_bf16(false, afC, false, bf2C, (short)0, a2, false, false);
    a3 = __builtin_amdgcn_wmma_f32_16x16x32_bf16(false, afC, false, bf3C, (short)0, a3, false, false);

    afC = afN; bf0C = bf0N; bf1C = bf1N; bf2C = bf2N; bf3C = bf3N;
  }
  // epilogue: last k-tile
  a0 = __builtin_amdgcn_wmma_f32_16x16x32_bf16(false, afC, false, bf0C, (short)0, a0, false, false);
  a1 = __builtin_amdgcn_wmma_f32_16x16x32_bf16(false, afC, false, bf1C, (short)0, a1, false, false);
  a2 = __builtin_amdgcn_wmma_f32_16x16x32_bf16(false, afC, false, bf2C, (short)0, a2, false, false);
  a3 = __builtin_amdgcn_wmma_f32_16x16x32_bf16(false, afC, false, bf3C, (short)0, a3, false, false);

  // C/D layout: lane holds column N=lane%16; VGPR r -> row M = r + (lane<16?0:8)
  int mrow = mt * 16 + ((lane < 16) ? 0 : 8);
  int ncol = lane & 15;
  float* hb = H + (size_t)b * (size_t)(Mtiles * 16) * N_;
#pragma unroll
  for (int t = 0; t < 4; ++t) {
    v8f acc = (t == 0) ? a0 : (t == 1) ? a1 : (t == 2) ? a2 : a3;
    int nc = (ng * 4 + t) * 16 + ncol;
#pragma unroll
    for (int rr = 0; rr < 8; ++rr) {
      hb[(size_t)(mrow + rr) * N_ + nc] = acc[rr];
    }
  }
}

// ---------------------------------------------------------------------------
// Training-mode BN stats per channel over (B,N): one block per channel,
// deterministic tree reduction. stats[c] = gamma*rsqrt(var+eps),
// stats[C+c] = beta - mean*scale.   (Conv bias cancels in BN -> ignored.)
// ---------------------------------------------------------------------------
__global__ __launch_bounds__(256) void bn_stats_kernel(
    const float* __restrict__ H, const float* __restrict__ gamma,
    const float* __restrict__ beta, float* __restrict__ stats, int C)
{
  __shared__ float ssum[256], ssq[256];
  int c = blockIdx.x;
  int tid = threadIdx.x;
  float s = 0.f, s2 = 0.f;
  const size_t cn = (size_t)C * N_;
  for (int i = tid; i < B_ * N_; i += 256) {
    int bb = i >> 12;           // / 4096
    int n  = i & (N_ - 1);
    float v = H[(size_t)bb * cn + (size_t)c * N_ + n];
    s += v; s2 += v * v;
  }
  ssum[tid] = s; ssq[tid] = s2;
  __syncthreads();
  for (int off = 128; off > 0; off >>= 1) {
    if (tid < off) { ssum[tid] += ssum[tid + off]; ssq[tid] += ssq[tid + off]; }
    __syncthreads();
  }
  if (tid == 0) {
    float inv = 1.0f / (float)(B_ * N_);
    float mean = ssum[0] * inv;
    float var  = ssq[0] * inv - mean * mean;
    float sc = gamma[c] * rsqrtf(var + EPS_BN);
    stats[c] = sc;
    stats[C + c] = beta[c] - mean * sc;
  }
}

// ---------------------------------------------------------------------------
// BN + ReLU applied to h0, re-packed as bf16 B-fragments for GEMM1.
// ---------------------------------------------------------------------------
__global__ void bn_relu_pack_kernel(const float* __restrict__ H,
                                    const float* __restrict__ stats,
                                    __bf16* __restrict__ Xp, int C)
{
  const int Ntiles = N_ / 16;
  int Ktiles = C / 32;
  int tid = blockIdx.x * blockDim.x + threadIdx.x;
  int total = B_ * Ntiles * Ktiles * 32;
  if (tid >= total) return;
  int lane = tid & 31;
  int kt = (tid >> 5) % Ktiles;
  int nt = ((tid >> 5) / Ktiles) % Ntiles;
  int b  = (tid >> 5) / (Ktiles * Ntiles);
  int n = nt * 16 + (lane & 15);
  int kbase = kt * 32 + ((lane < 16) ? 0 : 16);
  const float* hb = H + (size_t)b * C * N_;
  v16bf frag;
#pragma unroll
  for (int e = 0; e < 16; ++e) {
    int c = kbase + e;
    float v = hb[(size_t)c * N_ + n];
    frag[e] = f2bf(fmaxf(v * stats[c] + stats[C + c], 0.0f));
  }
  *(v16bf*)(Xp + (size_t)tid * 16) = frag;
}

// ---------------------------------------------------------------------------
// Final BN + ReLU -> f32 output [B][C1][N].
// ---------------------------------------------------------------------------
__global__ void bn_relu_out_kernel(const float* __restrict__ H,
                                   const float* __restrict__ stats,
                                   float* __restrict__ out)
{
  int tid = blockIdx.x * blockDim.x + threadIdx.x;
  const int total = B_ * C1_ * N_;
  if (tid >= total) return;
  int c = (tid >> 12) & (C1_ - 1);   // (tid / N_) % C1_
  out[tid] = fmaxf(H[tid] * stats[c] + stats[C1_ + c], 0.0f);
}

// ---------------------------------------------------------------------------
extern "C" void kernel_launch(void* const* d_in, const int* in_sizes, int n_in,
                              void* d_out, int out_size, void* d_ws, size_t ws_size,
                              hipStream_t stream)
{
  (void)in_sizes; (void)n_in; (void)out_size; (void)ws_size;
  const float* xyz1    = (const float*)d_in[0];
  const float* xyz2    = (const float*)d_in[1];
  const float* points1 = (const float*)d_in[2];
  const float* points2 = (const float*)d_in[3];
  const float* w0      = (const float*)d_in[4];
  /* b0 = d_in[5]: per-channel bias cancels exactly in training-mode BN */
  const float* g0      = (const float*)d_in[6];
  const float* be0     = (const float*)d_in[7];
  const float* w1      = (const float*)d_in[8];
  /* b1 = d_in[9]: cancels */
  const float* g1      = (const float*)d_in[10];
  const float* be1     = (const float*)d_in[11];

  char* p = (char*)d_ws;
  auto alloc = [&](size_t bytes) {
    char* r = p; p += (bytes + 255) & ~(size_t)255; return (void*)r;
  };
  int*    idx  = (int*)   alloc((size_t)B_ * N_ * 3 * sizeof(int));
  float*  wgt  = (float*) alloc((size_t)B_ * N_ * 3 * sizeof(float));
  __bf16* w0p  = (__bf16*)alloc((size_t)C0_ * CIN_ * sizeof(__bf16));
  __bf16* w1p  = (__bf16*)alloc((size_t)C1_ * C0_ * sizeof(__bf16));
  __bf16* xp   = (__bf16*)alloc((size_t)B_ * CIN_ * N_ * sizeof(__bf16));  // 50.3 MB
  float*  h0   = (float*) alloc((size_t)B_ * C0_ * N_ * sizeof(float));    // 67.1 MB
  float*  st0  = (float*) alloc(2 * C0_ * sizeof(float));
  float*  st1  = (float*) alloc(2 * C1_ * sizeof(float));
  __bf16* x1p  = xp;   // alias: xp dead after GEMM0
  float*  h1   = h0;   // alias: h0 dead after bn_relu_pack

  // 1) 3-NN + weights
  knn_kernel<<<B_ * (N_ / 256), 256, 0, stream>>>(xyz1, xyz2, idx, wgt);

  // 2) pack weights to A-fragment order
  {
    int t0 = (C0_ / 16) * (CIN_ / 32) * 32;
    pack_w_kernel<<<(t0 + 255) / 256, 256, 0, stream>>>(w0, w0p, C0_ / 16, CIN_ / 32, CIN_);
    int t1 = (C1_ / 16) * (C0_ / 32) * 32;
    pack_w_kernel<<<(t1 + 255) / 256, 256, 0, stream>>>(w1, w1p, C1_ / 16, C0_ / 32, C0_);
  }

  // 3) build GEMM0 input (concat + interpolation fused), bf16 B-fragments
  {
    int t = B_ * (N_ / 16) * (CIN_ / 32) * 32;
    pack_x0_kernel<<<(t + 255) / 256, 256, 0, stream>>>(points1, points2, idx, wgt, xp);
  }

  // 4) GEMM0: h0 = w0 @ x   (256 x 4096 per batch, K=384)
  {
    int waves = B_ * (C0_ / 16) * (N_ / 64);   // 16384
    gemm_wmma_kernel<<<waves / 8, 256, 0, stream>>>(w0p, xp, h0, C0_ / 16, CIN_ / 32);
  }

  // 5) BN stats + apply(BN,ReLU) fused with bf16 re-pack for GEMM1
  bn_stats_kernel<<<C0_, 256, 0, stream>>>(h0, g0, be0, st0, C0_);
  {
    int t = B_ * (N_ / 16) * (C0_ / 32) * 32;
    bn_relu_pack_kernel<<<(t + 255) / 256, 256, 0, stream>>>(h0, st0, x1p, C0_);
  }

  // 6) GEMM1: h1 = w1 @ relu(bn(h0))   (128 x 4096 per batch, K=256)
  {
    int waves = B_ * (C1_ / 16) * (N_ / 64);   // 8192
    gemm_wmma_kernel<<<waves / 8, 256, 0, stream>>>(w1p, x1p, h1, C1_ / 16, C0_ / 32);
  }

  // 7) BN stats + final apply -> d_out
  bn_stats_kernel<<<C1_, 256, 0, stream>>>(h1, g1, be1, st1, C1_);
  {
    int t = B_ * C1_ * N_;
    bn_relu_out_kernel<<<(t + 255) / 256, 256, 0, stream>>>(h1, st1, (float*)d_out);
  }
}
